// Encoder_6339371729763
// MI455X (gfx1250) — matrix operand order repaired and checked by reading, not compiled, not run on hardware
//
#include <hip/hip_runtime.h>

// ---------------------------------------------------------------------------
// CDNA5 (gfx1250) wave32 WMMA encoder.
// All GEMMs: A [M,K] bf16 row-major, B [N,K] bf16 row-major (pre-transposed
// weights), f32 accumulate.  Tile staging is pure async-DMA to LDS with
// double buffering (ASYNCcnt), fragments read with ds_load_b128.
// ---------------------------------------------------------------------------

typedef __attribute__((ext_vector_type(16))) __bf16 v16bf;
typedef __attribute__((ext_vector_type(8)))  float  v8f;

#define LDSS 40   // LDS row stride in bf16 elements (80B: dword + 16B aligned)

union FragBF { v16bf v; unsigned int u[8]; };

// Async DMA 16B global -> LDS (ASYNCcnt-tracked, no VGPR data round trip).
// Generic shared pointer's low 32 bits are the LDS byte offset.
__device__ __forceinline__ void async_copy_b128(const void* g, const void* l)
{
    unsigned loff = (unsigned)(unsigned long long)l;
    asm volatile("global_load_async_to_lds_b128 %0, %1, off"
                 :: "v"(loff), "v"(g) : "memory");
}
__device__ __forceinline__ void wait_async0()
{
    asm volatile("s_wait_asynccnt 0x0" ::: "memory");
}

// Tiled WMMA GEMM: C[MxN] = act( scale * A @ B^T + bias ),  B stored [N,K].
//   BN     : block tile N (64 or 128); block tile M fixed 64; K-step 32
//   RELU / BIAS : epilogue math
//   WF32 / WBF16: write f32 / bf16 output (row-major, ldc)
//   WTRANS : bf16 output stored transposed: Cb[col*ldc + row]
// 256 threads = 8 waves.
//   BN=64 : wave grid 4(M)x2(N), wave tile 16x32 (1 A-frag, 2 accs)
//   BN=128: wave grid 2(M)x4(N), wave tile 32x32 (2 A-frags, 4 accs)
template<int BN, bool RELU, bool BIAS, bool WF32, bool WBF16, bool WTRANS>
__global__ __launch_bounds__(256)
void gemm_wmma(const __bf16* __restrict__ A, long long sAb, int lda,
               const __bf16* __restrict__ Bm, long long sBb, int ldb,
               const float* __restrict__ bias, float scale,
               float* __restrict__ Cf, __bf16* __restrict__ Cb,
               long long sCb, int ldc, int K)
{
    constexpr int NW = BN / 32;   // waves along N
    constexpr int MW = 8 / NW;    // waves along M
    constexpr int WM = 64 / MW;   // rows per wave
    constexpr int AF = WM / 16;   // A fragments per wave
    constexpr int BR = BN / 64;   // 64-row blocks in the B tile

    __shared__ __bf16 Als[2][64 * LDSS];
    __shared__ __bf16 Bls[2][BN * LDSS];

    const int t    = threadIdx.x;
    const int lane = t & 31;
    const int wave = t >> 5;
    const int hl   = lane >> 4;          // lane half (K-half selector)
    const int m0   = blockIdx.y * 64;
    const int n0   = blockIdx.x * BN;
    const int z    = blockIdx.z;

    A  += (long long)z * sAb;
    Bm += (long long)z * sBb;

    const int mb = (wave % MW) * WM;
    const int nb = (wave / MW) * 32;

    v8f acc[AF][2];
    #pragma unroll
    for (int f = 0; f < AF; ++f) { acc[f][0] = (v8f){}; acc[f][1] = (v8f){}; }

    const int arow = t >> 2, aseg = (t & 3) * 8;   // 64 rows x 4 segs of 8 bf16

    // stage one K-tile (A 64x32 and B BNx32, both row-major [*,K]) via async DMA
    auto stage = [&](int buf, int k0) {
        async_copy_b128(A + (size_t)(m0 + arow) * lda + k0 + aseg,
                        &Als[buf][arow * LDSS + aseg]);
        #pragma unroll
        for (int r = 0; r < BR; ++r)
            async_copy_b128(Bm + (size_t)(n0 + r * 64 + arow) * ldb + k0 + aseg,
                            &Bls[buf][(r * 64 + arow) * LDSS + aseg]);
    };

    stage(0, 0);
    wait_async0();
    __syncthreads();

    int cur = 0;
    for (int k0 = 0; k0 < K; k0 += 32) {
        if (k0 + 32 < K) stage(cur ^ 1, k0 + 32);   // prefetch next tile

        // ---- fragments per ISA 16-bit layouts, from current buffer
        FragBF af[AF], bfr[2];
        const int ml = lane & 15;
        #pragma unroll
        for (int i = 0; i < 8; ++i) {
            // A 16x32: dword i holds K = (i>>2)*16 + half*8 + (i&3)*2, +1
            int ka = (i >> 2) * 16 + hl * 8 + (i & 3) * 2;
            #pragma unroll
            for (int f = 0; f < AF; ++f)
                af[f].u[i] = *(const unsigned int*)(&Als[cur][(mb + f * 16 + ml) * LDSS + ka]);
            // B 32x16: dword i holds K = (i>>2)*16 + half*8 + (i&3)*2, +1 for column N=lane&15
            int kb = (i >> 2) * 16 + hl * 8 + (i & 3) * 2;
            bfr[0].u[i] = *(const unsigned int*)(&Bls[cur][(nb + ml) * LDSS + kb]);
            bfr[1].u[i] = *(const unsigned int*)(&Bls[cur][(nb + 16 + ml) * LDSS + kb]);
        }
        #pragma unroll
        for (int f = 0; f < AF; ++f) {
            acc[f][0] = __builtin_amdgcn_wmma_f32_16x16x32_bf16(
                false, af[f].v, false, bfr[0].v, (short)0, acc[f][0], false, false);
            acc[f][1] = __builtin_amdgcn_wmma_f32_16x16x32_bf16(
                false, af[f].v, false, bfr[1].v, (short)0, acc[f][1], false, false);
        }

        wait_async0();       // next buffer landed
        __syncthreads();     // everyone done reading cur / writing next
        cur ^= 1;
    }

    // ---- epilogue: D vgpr i = row (half*8 + i), col = lane&15
    const int col0 = n0 + nb + (lane & 15);
    #pragma unroll
    for (int f = 0; f < AF; ++f) {
        #pragma unroll
        for (int i = 0; i < 8; ++i) {
            int row = m0 + mb + f * 16 + hl * 8 + i;
            float v0 = acc[f][0][i] * scale;
            float v1 = acc[f][1][i] * scale;
            if (BIAS) { v0 += bias[col0]; v1 += bias[col0 + 16]; }
            if (RELU) { v0 = fmaxf(v0, 0.f); v1 = fmaxf(v1, 0.f); }
            if (WF32) {
                float* cf = Cf + (long long)z * sCb + (size_t)row * ldc;
                cf[col0] = v0; cf[col0 + 16] = v1;
            }
            if (WBF16) {
                __bf16* cb = Cb + (long long)z * sCb;
                if (WTRANS) {   // store as [N, Mtot]
                    cb[(size_t)col0 * ldc + row]        = (__bf16)v0;
                    cb[(size_t)(col0 + 16) * ldc + row] = (__bf16)v1;
                } else {
                    cb[(size_t)row * ldc + col0]      = (__bf16)v0;
                    cb[(size_t)row * ldc + col0 + 16] = (__bf16)v1;
                }
            }
        }
    }
}

// ---- row softmax over n=2048, f32 in -> bf16 probabilities out -------------
__global__ __launch_bounds__(256)
void softmax_rows(const float* __restrict__ Sc, __bf16* __restrict__ P, int n)
{
    __shared__ float red[256];
    const int row = blockIdx.x;
    const float* src = Sc + (size_t)row * n;
    __bf16* dst = P + (size_t)row * n;
    int t = threadIdx.x;
    float vals[8];
    float mx = -3.4e38f;
    #pragma unroll
    for (int j = 0; j < 8; ++j) { vals[j] = src[t + j * 256]; mx = fmaxf(mx, vals[j]); }
    red[t] = mx; __syncthreads();
    for (int s = 128; s > 0; s >>= 1) { if (t < s) red[t] = fmaxf(red[t], red[t + s]); __syncthreads(); }
    mx = red[0]; __syncthreads();
    float sum = 0.f;
    #pragma unroll
    for (int j = 0; j < 8; ++j) { vals[j] = expf(vals[j] - mx); sum += vals[j]; }
    red[t] = sum; __syncthreads();
    for (int s = 128; s > 0; s >>= 1) { if (t < s) red[t] += red[t + s]; __syncthreads(); }
    float inv = 1.0f / red[0];
    #pragma unroll
    for (int j = 0; j < 8; ++j) dst[t + j * 256] = (__bf16)(vals[j] * inv);
}

// ---- y = LayerNorm(x + res) over D=512; write f32 and bf16 -----------------
__global__ __launch_bounds__(256)
void add_ln(const float* __restrict__ x, const float* __restrict__ res,
            const float* __restrict__ g, const float* __restrict__ b,
            float* __restrict__ yf, __bf16* __restrict__ yb)
{
    __shared__ float red[256];
    const int row = blockIdx.x;
    int t = threadIdx.x;
    const float* xr = x  + (size_t)row * 512;
    const float* rr = res + (size_t)row * 512;
    float a0 = xr[t] + rr[t];
    float a1 = xr[t + 256] + rr[t + 256];
    red[t] = a0 + a1; __syncthreads();
    for (int s = 128; s > 0; s >>= 1) { if (t < s) red[t] += red[t + s]; __syncthreads(); }
    float mean = red[0] * (1.0f / 512.0f); __syncthreads();
    float d0 = a0 - mean, d1 = a1 - mean;
    red[t] = d0 * d0 + d1 * d1; __syncthreads();
    for (int s = 128; s > 0; s >>= 1) { if (t < s) red[t] += red[t + s]; __syncthreads(); }
    float rstd = rsqrtf(red[0] * (1.0f / 512.0f) + 1e-5f);
    float y0 = d0 * rstd * g[t] + b[t];
    float y1 = d1 * rstd * g[t + 256] + b[t + 256];
    float* yfr = yf + (size_t)row * 512;
    __bf16* ybr = yb + (size_t)row * 512;
    yfr[t] = y0; yfr[t + 256] = y1;
    ybr[t] = (__bf16)y0; ybr[t + 256] = (__bf16)y1;
}

// ---- out = X + PE, PE applies cos to even SEQUENCE pos, sin to odd ---------
__global__ __launch_bounds__(256)
void pe_add(const float* __restrict__ X, float* __restrict__ yf, __bf16* __restrict__ yb)
{
    int row = blockIdx.x;          // b*S + s
    int s   = row & 2047;          // S = 2048
    int t   = threadIdx.x;
    const float* xr = X + (size_t)row * 512;
    float*  yfr = yf + (size_t)row * 512;
    __bf16* ybr = yb + (size_t)row * 512;
    #pragma unroll
    for (int j = 0; j < 2; ++j) {
        int d = t + j * 256;
        float freq = powf(1e-4f, (float)(2 * (d >> 1)) * (1.0f / 512.0f));
        float ang  = (float)s * freq;
        float pe   = (s & 1) ? sinf(ang) : cosf(ang);
        float v = xr[d] + pe;
        yfr[d] = v; ybr[d] = (__bf16)v;
    }
}

// transpose-convert: src [R,C] f32 row-major -> dst [C,R] bf16 row-major
__global__ __launch_bounds__(256)
void cvt_bf16_t(const float* __restrict__ s, __bf16* __restrict__ d, int R, int C)
{
    int i = blockIdx.x * 256 + threadIdx.x;
    if (i >= R * C) return;
    int r = i / C, c = i - r * C;
    d[(size_t)c * R + r] = (__bf16)s[i];
}

// tile(a,H) @ Wo == a @ Wo_eff; store transposed: woeT[c,d] = sum_h Wo[h*64+d, c]
__global__ __launch_bounds__(256)
void wo_fold_t(const float* __restrict__ Wo, __bf16* __restrict__ woeT)
{
    int i = blockIdx.x * 256 + threadIdx.x;   // 64*512
    if (i >= 64 * 512) return;
    int dd = i >> 9, c = i & 511;
    float s = 0.f;
    #pragma unroll
    for (int h = 0; h < 8; ++h) s += Wo[(size_t)(h * 64 + dd) * 512 + c];
    woeT[(size_t)c * 64 + dd] = (__bf16)s;
}

// ---------------------------------------------------------------------------
extern "C" void kernel_launch(void* const* d_in, const int* in_sizes, int n_in,
                              void* d_out, int out_size, void* d_ws, size_t ws_size,
                              hipStream_t stream)
{
    constexpr int BSZ = 4, SEQ = 2048, DIM = 512, HDIM = 64, FDIM = 2048;
    constexpr int MTOT = BSZ * SEQ;          // 8192
    constexpr int LAYERS = 6;

    const float* X   = (const float*)d_in[0];
    const float* Wq  = (const float*)d_in[1];
    const float* bq  = (const float*)d_in[2];
    const float* Wk  = (const float*)d_in[3];
    const float* bk  = (const float*)d_in[4];
    const float* Wv  = (const float*)d_in[5];
    const float* bv  = (const float*)d_in[6];
    const float* Wo  = (const float*)d_in[7];
    const float* bo  = (const float*)d_in[8];
    const float* Wf1 = (const float*)d_in[9];
    const float* bf1 = (const float*)d_in[10];
    const float* Wf2 = (const float*)d_in[11];
    const float* bf2 = (const float*)d_in[12];
    const float* g1  = (const float*)d_in[13];
    const float* b1  = (const float*)d_in[14];
    const float* g2  = (const float*)d_in[15];
    const float* b2  = (const float*)d_in[16];

    // workspace carve-up
    char* base = (char*)d_ws;
    size_t off = 0;
    auto alloc = [&](size_t bytes) -> void* {
        void* p = base + off;
        off = (off + bytes + 255) & ~(size_t)255;
        return p;
    };
    float*  outF   = (float*) alloc((size_t)MTOT * DIM * 4);
    float*  attF   = (float*) alloc((size_t)MTOT * DIM * 4);
    float*  projF  = (float*) alloc((size_t)MTOT * DIM * 4);
    __bf16* outB   = (__bf16*)alloc((size_t)MTOT * DIM * 2);
    __bf16* attB   = (__bf16*)alloc((size_t)MTOT * DIM * 2);
    __bf16* qB     = (__bf16*)alloc((size_t)MTOT * HDIM * 2);
    __bf16* kB     = (__bf16*)alloc((size_t)MTOT * HDIM * 2);
    __bf16* vT     = (__bf16*)alloc((size_t)HDIM * MTOT * 2);   // [64, B*S]
    __bf16* aB     = (__bf16*)alloc((size_t)MTOT * HDIM * 2);
    __bf16* hB     = (__bf16*)alloc((size_t)MTOT * FDIM * 2);
    float*  scoresF= (float*) alloc((size_t)BSZ * SEQ * SEQ * 4);
    __bf16* pB     = (__bf16*)alloc((size_t)BSZ * SEQ * SEQ * 2);
    __bf16* wqT    = (__bf16*)alloc((size_t)HDIM * DIM * 2);    // [64, 512]
    __bf16* wkT    = (__bf16*)alloc((size_t)HDIM * DIM * 2);
    __bf16* wvT    = (__bf16*)alloc((size_t)HDIM * DIM * 2);
    __bf16* woeT   = (__bf16*)alloc((size_t)DIM * HDIM * 2);    // [512, 64]
    __bf16* wf1T   = (__bf16*)alloc((size_t)FDIM * DIM * 2);    // [2048, 512]
    __bf16* wf2T   = (__bf16*)alloc((size_t)DIM * FDIM * 2);    // [512, 2048]
    (void)ws_size; (void)in_sizes; (void)n_in; (void)out_size;

    dim3 blk(256);

    // weight prep: bf16 + transpose to [N,K] (re-run every call; deterministic)
    cvt_bf16_t<<<dim3((DIM*HDIM + 255)/256), blk, 0, stream>>>(Wq, wqT, DIM, HDIM);
    cvt_bf16_t<<<dim3((DIM*HDIM + 255)/256), blk, 0, stream>>>(Wk, wkT, DIM, HDIM);
    cvt_bf16_t<<<dim3((DIM*HDIM + 255)/256), blk, 0, stream>>>(Wv, wvT, DIM, HDIM);
    cvt_bf16_t<<<dim3((DIM*FDIM + 255)/256), blk, 0, stream>>>(Wf1, wf1T, DIM, FDIM);
    cvt_bf16_t<<<dim3((FDIM*DIM + 255)/256), blk, 0, stream>>>(Wf2, wf2T, FDIM, DIM);
    wo_fold_t<<<dim3((HDIM*DIM + 255)/256), blk, 0, stream>>>(Wo, woeT);

    pe_add<<<dim3(MTOT), blk, 0, stream>>>(X, outF, outB);

    const float iscale = 0.125f;   // 1/sqrt(64)
    for (int l = 0; l < LAYERS; ++l) {
        // Q,K: [8192,512] @ wT[64,512]^T + b -> bf16 row-major [8192,64]
        gemm_wmma<64,false,true,false,true,false><<<dim3(1,128,1), blk, 0, stream>>>(
            outB, 0, DIM, wqT, 0, DIM, bq, 1.0f, nullptr, qB, 0, HDIM, DIM);
        gemm_wmma<64,false,true,false,true,false><<<dim3(1,128,1), blk, 0, stream>>>(
            outB, 0, DIM, wkT, 0, DIM, bk, 1.0f, nullptr, kB, 0, HDIM, DIM);
        // V: same GEMM, stored transposed as vT [64, 8192] for the P@V pass
        gemm_wmma<64,false,true,false,true,true><<<dim3(1,128,1), blk, 0, stream>>>(
            outB, 0, DIM, wvT, 0, DIM, bv, 1.0f, nullptr, vT, 0, MTOT, DIM);

        // scores = (Q @ K^T)/8 per batch: K is [S,64] = [N,K] already
        gemm_wmma<128,false,false,true,false,false><<<dim3(16,32,BSZ), blk, 0, stream>>>(
            qB, (long long)SEQ*HDIM, HDIM, kB, (long long)SEQ*HDIM, HDIM,
            nullptr, iscale, scoresF, nullptr, (long long)SEQ*SEQ, SEQ, HDIM);

        softmax_rows<<<dim3(BSZ*SEQ), blk, 0, stream>>>(scoresF, pB, SEQ);

        // a = P @ V per batch: B = vT[64, MTOT] (batch = column slice of SEQ)
        gemm_wmma<64,false,false,false,true,false><<<dim3(1,32,BSZ), blk, 0, stream>>>(
            pB, (long long)SEQ*SEQ, SEQ, vT, (long long)SEQ, MTOT,
            nullptr, 1.0f, nullptr, aB, (long long)SEQ*HDIM, HDIM, SEQ);

        // proj = a @ Wo_eff + bo : [8192,64] @ woeT[512,64]^T -> f32
        gemm_wmma<128,false,true,true,false,false><<<dim3(4,128,1), blk, 0, stream>>>(
            aB, 0, HDIM, woeT, 0, HDIM, bo, 1.0f, projF, nullptr, 0, DIM, HDIM);

        add_ln<<<dim3(MTOT), blk, 0, stream>>>(projF, outF, g1, b1, attF, attB);

        // h = relu(att @ Wf1 + bf1) : wf1T [2048,512]
        gemm_wmma<128,true,true,false,true,false><<<dim3(16,128,1), blk, 0, stream>>>(
            attB, 0, DIM, wf1T, 0, DIM, bf1, 1.0f, nullptr, hB, 0, FDIM, DIM);

        // proj = h @ Wf2 + bf2 : wf2T [512,2048]
        gemm_wmma<128,false,true,true,false,false><<<dim3(4,128,1), blk, 0, stream>>>(
            hB, 0, FDIM, wf2T, 0, FDIM, bf2, 1.0f, projF, nullptr, 0, DIM, FDIM);

        float* dstF = (l == LAYERS - 1) ? (float*)d_out : outF;
        add_ln<<<dim3(MTOT), blk, 0, stream>>>(projF, attF, g2, b2, dstF, outB);
    }
}
